// PyG_PointNet2_Alpha_Predictor_19189913878636
// MI455X (gfx1250) — matrix operand order repaired and checked by reading, not compiled, not run on hardware
//
#include <hip/hip_runtime.h>
#include <hip/hip_bf16.h>

typedef __attribute__((ext_vector_type(16))) _Float16 v16h;
typedef __attribute__((ext_vector_type(8)))  _Float16 v8h;
typedef __attribute__((ext_vector_type(8)))  float    v8f;
typedef __attribute__((ext_vector_type(4)))  int      v4i;

#define TILE_M 128
#define TILE_N 64
#define TILE_K 32
#define AS_STRIDE 40   // halfs: 80B row stride -> every 8-half chunk is 16B aligned
#define BS_STRIDE 40

// ---------------------------------------------------------------------------
// async global->LDS 16-byte copy (CDNA5 GLOBAL_LOAD_ASYNC_TO_LDS_B128)
// builtin signature (from hipcc diagnostic): (v4i* AS1, v4i* AS3, imm, cpol)
// ---------------------------------------------------------------------------
__device__ __forceinline__ void async_cp16(const _Float16* gsrc, _Float16* ldst)
{
#if __has_builtin(__builtin_amdgcn_global_load_async_to_lds_b128)
  __builtin_amdgcn_global_load_async_to_lds_b128(
      (__attribute__((address_space(1))) v4i*)gsrc,
      (__attribute__((address_space(3))) v4i*)ldst, 0, 0);
#else
  unsigned lds_addr =
      (unsigned)(size_t)(__attribute__((address_space(3))) void*)ldst;
  asm volatile("global_load_async_to_lds_b128 %0, %1, off"
               :: "v"(lds_addr), "v"(gsrc) : "memory");
#endif
}

__device__ __forceinline__ void async_wait_all()
{
  asm volatile("s_wait_asynccnt 0x0" ::: "memory");
}

// ---------------------------------------------------------------------------
// Fused GEMM: C[M,Nout] = act( [A1 | A2] @ W + bias ) in f16 with f32 acc.
// A1:[M,K1] A2:[M,K2] f16 row-major; Wt:[Nout][Ktot] f16 (pre-transposed).
// Block: 128 threads (4 waves). Block tile 128x64, K-step 32.
// Wave w owns rows [w*32, w*32+32): 2 A-frags x 4 B-frags = 8 wmma / K-step.
// Aligned tiles staged via GLOBAL_LOAD_ASYNC_TO_LDS_B128 (ASYNCcnt);
// ragged tiles (K edges / concat boundary / Nout==1) via guarded f16 loads.
// Requires M % 128 == 0 (all launches satisfy this).
// ---------------------------------------------------------------------------
__global__ __launch_bounds__(128) void gemm_cat_bias_act(
    const _Float16* __restrict__ A1, const _Float16* __restrict__ A2,
    const _Float16* __restrict__ Wt, const float* __restrict__ bias,
    _Float16* __restrict__ C, int M, int Nout, int K1, int K2, int relu)
{
  const int Ktot = K1 + K2;
  __shared__ _Float16 As[TILE_M][AS_STRIDE];
  __shared__ _Float16 Bs[TILE_N][BS_STRIDE];

  const int tid  = threadIdx.x;
  const int lane = tid & 31;
  const int wave = tid >> 5;
  const int m0 = blockIdx.x * TILE_M;
  const int n0 = blockIdx.y * TILE_N;

  const int nrow = lane & 15;   // M within 16-tile (A) / N within 16-tile (B, C/D)
  const int hsel = lane >> 4;   // half-wave select

  v8f acc[2][4] = {};

  for (int k0 = 0; k0 < Ktot; k0 += TILE_K) {
    // ========== stage A tile (128 rows x 32 halfs = 8KB) ==========
    if (k0 + TILE_K <= K1 && (K1 & 7) == 0) {
      // async path inside A1: 512 x b128, 4 per thread
      #pragma unroll
      for (int t = 0; t < 4; ++t) {
        int i = t * 128 + tid;
        int r = i >> 2, ch = i & 3;
        async_cp16(&A1[(size_t)(m0 + r) * K1 + k0 + ch * 8], &As[r][ch * 8]);
      }
    } else if (k0 >= K1 && k0 + TILE_K <= Ktot && (K2 & 7) == 0) {
      // async path inside A2
      const int kb = k0 - K1;
      #pragma unroll
      for (int t = 0; t < 4; ++t) {
        int i = t * 128 + tid;
        int r = i >> 2, ch = i & 3;
        async_cp16(&A2[(size_t)(m0 + r) * K2 + kb + ch * 8], &As[r][ch * 8]);
      }
    } else {
      // ragged path: guarded scalar f16 loads, 32 per thread
      #pragma unroll
      for (int t = 0; t < 32; ++t) {
        int i = t * 128 + tid;
        int r = i >> 5, c = i & 31;
        int gk = k0 + c;
        _Float16 v = (_Float16)0.0f;
        if (gk < Ktot) {
          if (gk < K1) v = A1[(size_t)(m0 + r) * K1 + gk];
          else         v = A2[(size_t)(m0 + r) * K2 + (gk - K1)];
        }
        As[r][c] = v;
      }
    }

    // ========== stage B tile: Bs[n][k] from Wt[Nout][Ktot] rows ==========
    if (k0 + TILE_K <= Ktot && (Ktot & 7) == 0 && n0 + TILE_N <= Nout) {
      // async path: 256 x b128, 2 per thread (Wt rows are contiguous K runs)
      #pragma unroll
      for (int t = 0; t < 2; ++t) {
        int i = t * 128 + tid;
        int n = i >> 2, ch = i & 3;
        async_cp16(&Wt[(size_t)(n0 + n) * Ktot + k0 + ch * 8], &Bs[n][ch * 8]);
      }
    } else {
      #pragma unroll
      for (int t = 0; t < 16; ++t) {
        int i = t * 128 + tid;
        int n = i >> 5, c = i & 31;   // 64 rows x 32 halfs
        int gk = k0 + c, gn = n0 + n;
        _Float16 v = (_Float16)0.0f;
        if (gk < Ktot && gn < Nout) v = Wt[(size_t)gn * Ktot + gk];
        Bs[n][c] = v;
      }
    }

    async_wait_all();
    __syncthreads();

    // ========== fragments (ds_load_b128 pairs) ==========
    // A lane layout: M = lane%16, halfs = K in {h*8..h*8+7, 16+h*8..16+h*8+7}
    v8h a0lo = *(const v8h*)&As[wave * 32 + nrow     ][hsel * 8];
    v8h a0hi = *(const v8h*)&As[wave * 32 + nrow     ][16 + hsel * 8];
    v8h a1lo = *(const v8h*)&As[wave * 32 + 16 + nrow][hsel * 8];
    v8h a1hi = *(const v8h*)&As[wave * 32 + 16 + nrow][16 + hsel * 8];
    v16h afr[2];
    afr[0] = __builtin_shufflevector(a0lo, a0hi, 0,1,2,3,4,5,6,7,8,9,10,11,12,13,14,15);
    afr[1] = __builtin_shufflevector(a1lo, a1hi, 0,1,2,3,4,5,6,7,8,9,10,11,12,13,14,15);

    // B lane layout: N = lane%16, halfs = K in {h*8..h*8+7, 16+h*8..16+h*8+7}
    v16h bfr[4];
    #pragma unroll
    for (int nt = 0; nt < 4; ++nt) {
      v8h blo = *(const v8h*)&Bs[nt * 16 + nrow][hsel * 8];
      v8h bhi = *(const v8h*)&Bs[nt * 16 + nrow][16 + hsel * 8];
      bfr[nt] = __builtin_shufflevector(blo, bhi, 0,1,2,3,4,5,6,7,8,9,10,11,12,13,14,15);
    }

    #pragma unroll
    for (int mt = 0; mt < 2; ++mt) {
      #pragma unroll
      for (int nt = 0; nt < 4; ++nt) {
        acc[mt][nt] = __builtin_amdgcn_wmma_f32_16x16x32_f16(
            false, afr[mt], false, bfr[nt], (short)0, acc[mt][nt], false, false);
      }
    }
    __syncthreads();
  }

  // ========== epilogue: bias + act + f16 store ==========
  #pragma unroll
  for (int mt = 0; mt < 2; ++mt) {
    #pragma unroll
    for (int r = 0; r < 8; ++r) {
      const int gr = m0 + wave * 32 + mt * 16 + r + 8 * hsel;  // M % 128 == 0
      #pragma unroll
      for (int nt = 0; nt < 4; ++nt) {
        const int gn = n0 + nt * 16 + nrow;
        if (gn < Nout) {
          float val = acc[mt][nt][r] + bias[gn];
          if (relu) val = fmaxf(val, 0.0f);
          C[(size_t)gr * Nout + gn] = (_Float16)val;
        }
      }
    }
  }
}

// ---------------------------------------------------------------------------
// weight convert + transpose: Wt[n][k] = (f16) W[k][n]
__global__ void convert_w_kernel(const float* __restrict__ W, _Float16* __restrict__ Wt,
                                 int K, int N)
{
  int i = blockIdx.x * blockDim.x + threadIdx.x;
  if (i >= K * N) return;
  int k = i / N, n = i % N;
  Wt[(size_t)n * K + k] = (_Float16)W[i];
}

__global__ void cvt_f16_kernel(const float* __restrict__ in, _Float16* __restrict__ out, int n)
{
  int t = blockIdx.x * blockDim.x + threadIdx.x;
  if (t < n) out[t] = (_Float16)in[t];
}

// ---------------------------------------------------------------------------
// Farthest point sampling (f32 geometry): 1 workgroup per cloud.
__global__ __launch_bounds__(1024) void fps_kernel(
    const float* __restrict__ pos, int n, int m, int* __restrict__ out_idx)
{
  __shared__ float mind[8192];
  __shared__ float sval[1024];
  __shared__ int   sidx[1024];
  const int b   = blockIdx.x;
  const int tid = threadIdx.x;
  const float* pb = pos + (size_t)b * n * 3;

  for (int i = tid; i < n; i += 1024) mind[i] = 1e10f;
  if (tid == 0) out_idx[(size_t)b * m] = 0;
  __syncthreads();

  int last = 0;
  for (int s = 1; s < m; ++s) {
    const float lx = pb[last * 3 + 0];
    const float ly = pb[last * 3 + 1];
    const float lz = pb[last * 3 + 2];
    float best = -1.0f; int bi = 0x7fffffff;
    for (int i = tid; i < n; i += 1024) {
      float dx = pb[i * 3 + 0] - lx;
      float dy = pb[i * 3 + 1] - ly;
      float dz = pb[i * 3 + 2] - lz;
      float d  = dx * dx + dy * dy + dz * dz;
      float mn = fminf(mind[i], d);
      mind[i] = mn;
      if (mn > best || (mn == best && i < bi)) { best = mn; bi = i; }
    }
    sval[tid] = best; sidx[tid] = bi;
    __syncthreads();
    for (int off2 = 512; off2 > 0; off2 >>= 1) {
      if (tid < off2) {
        float v2 = sval[tid + off2]; int i2 = sidx[tid + off2];
        if (v2 > sval[tid] || (v2 == sval[tid] && i2 < sidx[tid])) {
          sval[tid] = v2; sidx[tid] = i2;
        }
      }
      __syncthreads();
    }
    last = sidx[0];
    if (tid == 0) out_idx[(size_t)b * m + s] = last;
    __syncthreads();
  }
}

// dst[b,i,:] = src[b, idx[b,i], :]  (f32: positions)
__global__ void gather_rows_f32(const float* __restrict__ src, const int* __restrict__ idx,
                                float* __restrict__ dst, int n, int m, int C, int B)
{
  size_t t = (size_t)blockIdx.x * blockDim.x + threadIdx.x;
  size_t total = (size_t)B * m * C;
  if (t >= total) return;
  int c = (int)(t % C);
  size_t p = t / C;
  int i = (int)(p % m);
  int b = (int)(p / m);
  int row = idx[(size_t)b * m + i];
  dst[t] = src[((size_t)b * n + row) * C + c];
}

// dst[b,i,:] = src[b, idx[b,i], :]  (f16: features)
__global__ void gather_rows_f16(const _Float16* __restrict__ src, const int* __restrict__ idx,
                                _Float16* __restrict__ dst, int n, int m, int C, int B)
{
  size_t t = (size_t)blockIdx.x * blockDim.x + threadIdx.x;
  size_t total = (size_t)B * m * C;
  if (t >= total) return;
  int c = (int)(t % C);
  size_t p = t / C;
  int i = (int)(p % m);
  int b = (int)(p / m);
  int row = idx[(size_t)b * m + i];
  dst[t] = src[((size_t)b * n + row) * C + c];
}

// brute-force 3-NN with inverse-square-distance weights (f32 geometry)
__global__ void knn3_kernel(const float* __restrict__ pdst, const float* __restrict__ psrc,
                            int Nd, int Ns, int B,
                            int* __restrict__ kidx, float* __restrict__ kw)
{
  int t = blockIdx.x * blockDim.x + threadIdx.x;
  if (t >= B * Nd) return;
  int b = t / Nd;
  const float* pd = pdst + (size_t)t * 3;
  const float* ps = psrc + (size_t)b * Ns * 3;
  const float x = pd[0], y = pd[1], z = pd[2];
  float d0 = 3.4e38f, d1 = 3.4e38f, d2 = 3.4e38f;
  int   i0 = 0, i1 = 0, i2 = 0;
  for (int j = 0; j < Ns; ++j) {
    float dx = ps[j * 3 + 0] - x;
    float dy = ps[j * 3 + 1] - y;
    float dz = ps[j * 3 + 2] - z;
    float d = dx * dx + dy * dy + dz * dz;
    if (d < d0)      { d2 = d1; i2 = i1; d1 = d0; i1 = i0; d0 = d; i0 = j; }
    else if (d < d1) { d2 = d1; i2 = i1; d1 = d;  i1 = j; }
    else if (d < d2) { d2 = d;  i2 = j; }
  }
  kidx[(size_t)t * 3 + 0] = i0;
  kidx[(size_t)t * 3 + 1] = i1;
  kidx[(size_t)t * 3 + 2] = i2;
  kw[(size_t)t * 3 + 0] = 1.0f / fmaxf(d0, 1e-16f);
  kw[(size_t)t * 3 + 1] = 1.0f / fmaxf(d1, 1e-16f);
  kw[(size_t)t * 3 + 2] = 1.0f / fmaxf(d2, 1e-16f);
}

// out[b,p,c] = sum_j w_j * fsrc[b, idx_j, c] / sum_j w_j   (f16 features)
__global__ void interp3_kernel(const _Float16* __restrict__ fsrc, const int* __restrict__ kidx,
                               const float* __restrict__ kw, _Float16* __restrict__ out,
                               int Nd, int Ns, int C, int B)
{
  size_t t = (size_t)blockIdx.x * blockDim.x + threadIdx.x;
  size_t total = (size_t)B * Nd * C;
  if (t >= total) return;
  int c = (int)(t % C);
  size_t p = t / C;
  int b = (int)(p / Nd);
  const int*   id = kidx + p * 3;
  const float* w  = kw   + p * 3;
  const _Float16* base = fsrc + (size_t)b * Ns * C;
  float s = w[0] * (float)base[(size_t)id[0] * C + c]
          + w[1] * (float)base[(size_t)id[1] * C + c]
          + w[2] * (float)base[(size_t)id[2] * C + c];
  out[t] = (_Float16)(s / (w[0] + w[1] + w[2]));
}

__global__ void softplus_kernel(const _Float16* __restrict__ in, float* __restrict__ out, int n)
{
  int t = blockIdx.x * blockDim.x + threadIdx.x;
  if (t < n) {
    float x = (float)in[t];
    out[t] = fmaxf(x, 0.0f) + log1pf(expf(-fabsf(x)));
  }
}

// ---------------------------------------------------------------------------
extern "C" void kernel_launch(void* const* d_in, const int* in_sizes, int n_in,
                              void* d_out, int out_size, void* d_ws, size_t ws_size,
                              hipStream_t stream)
{
  (void)in_sizes; (void)n_in; (void)out_size; (void)ws_size;

  const int Bc = 16, Np = 8192, N1 = 2048, N2 = 512;
  const int M0 = Bc * Np, M1 = Bc * N1, M2 = Bc * N2;

  const float* pos = (const float*)d_in[0];
  auto F = [&](int i) { return (const float*)d_in[i]; };

  // ---- static bump allocation inside d_ws (deterministic every call)
  size_t off = 0;
  auto allocB = [&](size_t bytes) -> void* {
    void* p = (void*)((char*)d_ws + off);
    off += (bytes + 255) & ~(size_t)255;
    return p;
  };
  auto allocH = [&](size_t ne) -> _Float16* { return (_Float16*)allocB(ne * 2); };
  auto allocF = [&](size_t ne) -> float*    { return (float*)allocB(ne * 4); };
  auto allocI = [&](size_t ne) -> int*      { return (int*)allocB(ne * 4); };

  _Float16* pos_h  = allocH((size_t)M0 * 3);
  _Float16* f0     = allocH((size_t)M0 * 128);
  _Float16* f0g    = allocH((size_t)M1 * 128);
  float*    p1     = allocF((size_t)M1 * 3);
  _Float16* p1_h   = allocH((size_t)M1 * 3);
  _Float16* f1     = allocH((size_t)M1 * 256);
  _Float16* f1g    = allocH((size_t)M2 * 256);
  float*    p2     = allocF((size_t)M2 * 3);
  _Float16* p2_h   = allocH((size_t)M2 * 3);
  _Float16* f2     = allocH((size_t)M2 * 1024);
  _Float16* interp = allocH((size_t)M1 * 1024);  // i1 [M1,1024] then i0 [M0,256]
  _Float16* f1fp   = allocH((size_t)M1 * 256);
  _Float16* f0fp   = allocH((size_t)M0 * 128);
  _Float16* ff     = allocH((size_t)M0 * 128);
  _Float16* alpha  = allocH((size_t)M0);
  _Float16* tA     = allocH((size_t)M0 * 256);   // hidden scratch A
  _Float16* tB     = allocH((size_t)M0 * 128);   // hidden scratch B
  float*    kw     = allocF((size_t)M0 * 3);
  int*      idx1   = allocI((size_t)M1);
  int*      idx2   = allocI((size_t)M2);
  int*      kidx   = allocI((size_t)M0 * 3);

  // ---- convert + transpose all 19 weight matrices to f16 Wt[N][K]
  static const int LW[19][3] = {   // {d_in index of W, K, N}
    {1,3,64},{3,64,64},{5,64,128},
    {7,131,128},{9,128,128},{11,128,256},
    {13,259,256},{15,256,512},{17,512,1024},
    {19,1280,256},{21,256,256},
    {23,384,256},{25,256,128},
    {27,256,128},{29,128,128},{31,128,128},
    {33,131,128},{35,128,64},{37,64,1}};
  _Float16* Wt[19];
  for (int j = 0; j < 19; ++j) {
    int K = LW[j][1], N = LW[j][2];
    Wt[j] = allocH((size_t)K * N);
    int tot = K * N;
    convert_w_kernel<<<(tot + 255) / 256, 256, 0, stream>>>(F(LW[j][0]), Wt[j], K, N);
  }

  auto gemm = [&](const _Float16* A1, int K1, const _Float16* A2, int K2,
                  int layer, _Float16* C, int M, int Nout, int relu) {
    dim3 g((M + TILE_M - 1) / TILE_M, (Nout + TILE_N - 1) / TILE_N);
    gemm_cat_bias_act<<<g, 128, 0, stream>>>(A1, A2, Wt[layer], F(LW[layer][0] + 1),
                                             C, M, Nout, K1, K2, relu);
  };
  auto blocks = [](size_t total) { return (unsigned)((total + 255) / 256); };

  // ---- pos -> f16
  cvt_f16_kernel<<<blocks((size_t)M0 * 3), 256, 0, stream>>>(pos, pos_h, M0 * 3);

  // ---- sa1: pos -> f0   [3 -> 64 -> 64 -> 128]
  gemm(pos_h, 3, nullptr, 0, 0, tA, M0, 64, 1);
  gemm(tA, 64, nullptr, 0, 1, tB, M0, 64, 1);
  gemm(tB, 64, nullptr, 0, 2, f0, M0, 128, 0);

  // ---- FPS level 1 + gathers
  fps_kernel<<<Bc, 1024, 0, stream>>>(pos, Np, N1, idx1);
  gather_rows_f32<<<blocks((size_t)M1 * 3), 256, 0, stream>>>(pos, idx1, p1, Np, N1, 3, Bc);
  cvt_f16_kernel<<<blocks((size_t)M1 * 3), 256, 0, stream>>>(p1, p1_h, M1 * 3);
  gather_rows_f16<<<blocks((size_t)M1 * 128), 256, 0, stream>>>(f0, idx1, f0g, Np, N1, 128, Bc);

  // ---- sa2: [f0g | p1] -> f1   [131 -> 128 -> 128 -> 256]
  gemm(f0g, 128, p1_h, 3, 3, tA, M1, 128, 1);
  gemm(tA, 128, nullptr, 0, 4, tB, M1, 128, 1);
  gemm(tB, 128, nullptr, 0, 5, f1, M1, 256, 0);

  // ---- FPS level 2 + gathers
  fps_kernel<<<Bc, 1024, 0, stream>>>(p1, N1, N2, idx2);
  gather_rows_f32<<<blocks((size_t)M2 * 3), 256, 0, stream>>>(p1, idx2, p2, N1, N2, 3, Bc);
  cvt_f16_kernel<<<blocks((size_t)M2 * 3), 256, 0, stream>>>(p2, p2_h, M2 * 3);
  gather_rows_f16<<<blocks((size_t)M2 * 256), 256, 0, stream>>>(f1, idx2, f1g, N1, N2, 256, Bc);

  // ---- sa3: [f1g | p2] -> f2   [259 -> 256 -> 512 -> 1024]
  gemm(f1g, 256, p2_h, 3, 6, tA, M2, 256, 1);
  gemm(tA, 256, nullptr, 0, 7, tB, M2, 512, 1);
  gemm(tB, 512, nullptr, 0, 8, f2, M2, 1024, 0);

  // ---- FP: level2 -> level1  (i1 = knn_interp(f2, p2 -> p1))
  knn3_kernel<<<blocks((size_t)M1), 256, 0, stream>>>(p1, p2, N1, N2, Bc, kidx, kw);
  interp3_kernel<<<blocks((size_t)M1 * 1024), 256, 0, stream>>>(f2, kidx, kw, interp,
                                                                N1, N2, 1024, Bc);
  // fp3: [i1 | f1] -> f1fp   [1280 -> 256 -> 256]
  gemm(interp, 1024, f1, 256, 9, tA, M1, 256, 1);
  gemm(tA, 256, nullptr, 0, 10, f1fp, M1, 256, 0);

  // ---- FP: level1 -> level0  (i0 = knn_interp(f1fp, p1 -> pos))
  knn3_kernel<<<blocks((size_t)M0), 256, 0, stream>>>(pos, p1, Np, N1, Bc, kidx, kw);
  interp3_kernel<<<blocks((size_t)M0 * 256), 256, 0, stream>>>(f1fp, kidx, kw, interp,
                                                               Np, N1, 256, Bc);
  // fp2: [i0 | f0] -> f0fp   [384 -> 256 -> 128]
  gemm(interp, 256, f0, 128, 11, tA, M0, 256, 1);
  gemm(tA, 256, nullptr, 0, 12, f0fp, M0, 128, 0);

  // ---- fp1: [f0fp | f0] -> ff   [256 -> 128 -> 128 -> 128]
  gemm(f0fp, 128, f0, 128, 13, tA, M0, 128, 1);
  gemm(tA, 128, nullptr, 0, 14, tB, M0, 128, 1);
  gemm(tB, 128, nullptr, 0, 15, ff, M0, 128, 0);

  // ---- head: [ff | pos] -> alpha   [131 -> 128 -> 64 -> 1]
  gemm(ff, 128, pos_h, 3, 16, tA, M0, 128, 1);
  gemm(tA, 128, nullptr, 0, 17, tB, M0, 64, 1);
  gemm(tB, 64, nullptr, 0, 18, alpha, M0, 1, 0);

  // ---- softplus; [B,N,1] -> [B,1,N] transpose is a no-op in memory
  softplus_kernel<<<blocks((size_t)M0), 256, 0, stream>>>(alpha, (float*)d_out, M0);
}
